// Attention_455266533863
// MI455X (gfx1250) — matrix operand order repaired and checked by reading, not compiled, not run on hardware
//
#include <hip/hip_runtime.h>
#include <hip/hip_bf16.h>

// MI455X / gfx1250, wave32. All three contractions on v_wmma_f32_16x16x32_bf16.

typedef __attribute__((ext_vector_type(16))) __bf16 v16bf;
typedef __attribute__((ext_vector_type(8)))  __bf16 v8bf;
typedef __attribute__((ext_vector_type(8)))  float  v8f;
typedef __attribute__((ext_vector_type(4)))  float  v4f;

__device__ __forceinline__ v16bf bf_cat(v8bf lo, v8bf hi) {
  return __builtin_shufflevector(lo, hi, 0,1,2,3,4,5,6,7,8,9,10,11,12,13,14,15);
}

__device__ __forceinline__ v8bf cvt8(v4f f0, v4f f1) {
  v8bf d;
#pragma unroll
  for (int u = 0; u < 4; ++u) { d[u] = (__bf16)f0[u]; d[u+4] = (__bf16)f1[u]; }
  return d;
}

// ---------------------------------------------------------------------------
// Elementwise f32 -> bf16 (one-time conversion of k and W; pure bandwidth).
// ---------------------------------------------------------------------------
__global__ __launch_bounds__(256) void cvt_f32_to_bf16(
    const float* __restrict__ src, __bf16* __restrict__ dst, long long n) {
  const long long i = ((long long)blockIdx.x * 256 + threadIdx.x) * 8;
  if (i + 8 <= n) {
    v4f f0 = *(const v4f*)(src + i);
    v4f f1 = *(const v4f*)(src + i + 4);
    *(v8bf*)(dst + i) = cvt8(f0, f1);
  }
}

// ---------------------------------------------------------------------------
// Row-major GEMM  C[M,N] = A[M,K] * B[K,N], B always bf16.
// Workgroup tile 128x64, 8 waves; wave w owns rows 16w..16w+15, all 64 cols.
// K-step 32. Register double-buffer: next global tiles are loaded while the
// WMMAs consume the current LDS tiles.
// ---------------------------------------------------------------------------
template<bool A_F32, bool OUT_BF16>
__global__ __launch_bounds__(256) void gemm_wmma_128x64(
    const void* __restrict__ Ap, const __bf16* __restrict__ Bp,
    void* __restrict__ Cp,
    int M, int N, int K,
    long long aStride, long long bStride, long long cStride)
{
  constexpr int BM = 128, BN = 64, BK = 32;
  __shared__ __align__(16) __bf16 ldsA[BM * BK];   // [m][k]   8 KB
  __shared__ __align__(16) __bf16 ldsBt[BN * BK];  // [n][k]   4 KB

  const int z = blockIdx.z;
  const float*  Af = (const float*)Ap  + (A_F32 ? (long long)z * aStride : 0);
  const __bf16* Ab = (const __bf16*)Ap + (A_F32 ? 0 : (long long)z * aStride);
  const __bf16* Bb = Bp + (long long)z * bStride;

  const int row0 = blockIdx.y * BM;
  const int col0 = blockIdx.x * BN;
  const int t    = threadIdx.x;
  const int wave = t >> 5, lane = t & 31;
  const int half = lane >> 4, m16 = lane & 15;

  // Per-thread staging coordinates.
  const int arow_ld = t >> 2;         // A: rows 0..63 (+64 for second chunk)
  const int aoff_ld = (t & 3) * 8;    //    k offset 0/8/16/24
  const int bk_ld   = t >> 3;         // B: k row 0..31
  const int bn_ld   = (t & 7) * 8;    //    n offset 0..56

  v8bf aReg[2], bReg;

  auto loadTiles = [&](int k0) {
    if (A_F32) {
#pragma unroll
      for (int j = 0; j < 2; ++j) {
        const float* s = Af + (long long)(row0 + arow_ld + 64 * j) * K + k0 + aoff_ld;
        aReg[j] = cvt8(*(const v4f*)s, *(const v4f*)(s + 4));
      }
    } else {
#pragma unroll
      for (int j = 0; j < 2; ++j)
        aReg[j] = *(const v8bf*)(Ab + (long long)(row0 + arow_ld + 64 * j) * K + k0 + aoff_ld);
    }
    bReg = *(const v8bf*)(Bb + (long long)(k0 + bk_ld) * N + col0 + bn_ld);
  };
  auto storeTiles = [&]() {
#pragma unroll
    for (int j = 0; j < 2; ++j)
      *(v8bf*)&ldsA[(arow_ld + 64 * j) * BK + aoff_ld] = aReg[j];
#pragma unroll
    for (int u = 0; u < 8; ++u)            // transpose store -> [n][k]
      ldsBt[(bn_ld + u) * BK + bk_ld] = bReg[u];
  };

  v8f acc[4] = {};
  loadTiles(0);
  for (int k0 = 0; k0 < K; k0 += BK) {
    storeTiles();
    __syncthreads();
    if (k0 + BK < K) loadTiles(k0 + BK);   // overlap with WMMAs below

    // A fragment: lanes 0-15 hold K{0..15}, lanes 16-31 K{16..31}
    const int arow = wave * 16 + m16;
    v16bf a = bf_cat(*(const v8bf*)&ldsA[arow * BK + 16 * half],
                     *(const v8bf*)&ldsA[arow * BK + 16 * half + 8]);
#pragma unroll
    for (int j = 0; j < 4; ++j) {
      const int bn = j * 16 + m16;
      v16bf bb = bf_cat(*(const v8bf*)&ldsBt[bn * BK + 16 * half],
                        *(const v8bf*)&ldsBt[bn * BK + 16 * half + 8]);
      acc[j] = __builtin_amdgcn_wmma_f32_16x16x32_bf16(
          false, a, false, bb, (short)0, acc[j], false, false);
    }
    __syncthreads();
  }

  // epilogue: VGPR i, lanes 0-15 -> M=i, lanes 16-31 -> M=i+8; N = lane&15
#pragma unroll
  for (int j = 0; j < 4; ++j)
#pragma unroll
    for (int i = 0; i < 8; ++i) {
      const long long r = row0 + wave * 16 + i + 8 * half;
      const long long c = col0 + j * 16 + m16;
      if (OUT_BF16)
        ((__bf16*)Cp)[(long long)z * cStride + r * N + c] = (__bf16)acc[j][i];
      else
        ((float*)Cp)[(long long)z * cStride + r * N + c] = acc[j][i];
    }
}

// ---------------------------------------------------------------------------
// Fused: S = qw @ k^T (E=1024), softmax over KL=2048, post-softmax mask,
// write P (bf16). One workgroup = 16 q-rows of one batch. Wave w owns score
// columns [256w, 256w+256): 16 register-resident 16x16 f32 accumulators.
// k is pre-converted bf16, so B fragments are two raw global_load_b128
// (no VALU between load and WMMA -> no hazard NOPs in the hot loop).
// ---------------------------------------------------------------------------
__global__ __launch_bounds__(256) void score_softmax_16q(
    const __bf16* __restrict__ qw,   // [B*Q, E]
    const __bf16* __restrict__ kbf,  // [B, KL, E]
    const int*    __restrict__ mask, // [B, Q, KL]
    __bf16*       __restrict__ P)    // [B*Q, KL]
{
  constexpr int E = 1024, KL = 2048, QT = 16;
  __shared__ __align__(16) __bf16 ldsQ[QT * E];  // 32 KB
  __shared__ float redBuf[16 * 8];
  __shared__ float rowM[16];
  __shared__ float rowL[16];

  const int blk = blockIdx.x;               // 0..B*Q/16-1
  const int b   = blk >> 7;                 // Q/16 = 128 blocks per batch
  const int q0  = (blk & 127) * QT;
  const long long qrow0 = (long long)b * 2048 + q0;

  const int t = threadIdx.x, wave = t >> 5, lane = t & 31;
  const int half = lane >> 4, m16 = lane & 15;

  // stage the 16 qw rows (bf16) into LDS
  {
    const v8bf* src = (const v8bf*)(qw + qrow0 * E);
    v8bf* dst = (v8bf*)ldsQ;
    for (int i = t; i < QT * E / 8; i += 256) dst[i] = src[i];
  }
  __syncthreads();

  v8f acc[16] = {};
  const int key0 = wave * 256;
  const __bf16* kb = kbf + (long long)b * KL * E;

  for (int e = 0; e < E; e += 32) {
    v16bf a = bf_cat(*(const v8bf*)&ldsQ[m16 * E + e + 16 * half],
                     *(const v8bf*)&ldsQ[m16 * E + e + 16 * half + 8]);
#pragma unroll
    for (int kt = 0; kt < 16; ++kt) {
      // B = k^T : lane holds key column (key0+kt*16+m16), 16 contiguous e's
      const __bf16* src = kb + (long long)(key0 + kt * 16 + m16) * E + e + 16 * half;
      v16bf bb = bf_cat(*(const v8bf*)src, *(const v8bf*)(src + 8));
      acc[kt] = __builtin_amdgcn_wmma_f32_16x16x32_bf16(
          false, a, false, bb, (short)0, acc[kt], false, false);
    }
  }

  // ---- row max (softmax denominator is over ALL keys; mask comes after) ----
  v8f pm = acc[0];
#pragma unroll
  for (int kt = 1; kt < 16; ++kt)
#pragma unroll
    for (int i = 0; i < 8; ++i) pm[i] = fmaxf(pm[i], acc[kt][i]);
#pragma unroll
  for (int off = 8; off >= 1; off >>= 1)
#pragma unroll
    for (int i = 0; i < 8; ++i)
      pm[i] = fmaxf(pm[i], __shfl_xor(pm[i], off, 32));
  if (m16 == 0) {
#pragma unroll
    for (int i = 0; i < 8; ++i) redBuf[(i + 8 * half) * 8 + wave] = pm[i];
  }
  __syncthreads();
  if (t < 16) {
    float m = redBuf[t * 8];
    for (int wv = 1; wv < 8; ++wv) m = fmaxf(m, redBuf[t * 8 + wv]);
    rowM[t] = m;
  }
  __syncthreads();

  float mrow[8];
#pragma unroll
  for (int i = 0; i < 8; ++i) mrow[i] = rowM[i + 8 * half];

  // ---- exp + full-row sum ----
  float ps[8] = {0.f, 0.f, 0.f, 0.f, 0.f, 0.f, 0.f, 0.f};
#pragma unroll
  for (int kt = 0; kt < 16; ++kt)
#pragma unroll
    for (int i = 0; i < 8; ++i) {
      float ev = __expf(acc[kt][i] - mrow[i]);
      acc[kt][i] = ev;
      ps[i] += ev;
    }
#pragma unroll
  for (int off = 8; off >= 1; off >>= 1)
#pragma unroll
    for (int i = 0; i < 8; ++i) ps[i] += __shfl_xor(ps[i], off, 32);
  if (m16 == 0) {
#pragma unroll
    for (int i = 0; i < 8; ++i) redBuf[(i + 8 * half) * 8 + wave] = ps[i];
  }
  __syncthreads();
  if (t < 16) {
    float s = 0.f;
    for (int wv = 0; wv < 8; ++wv) s += redBuf[t * 8 + wv];
    rowL[t] = s;
  }
  __syncthreads();

  float inv[8];
#pragma unroll
  for (int i = 0; i < 8; ++i) inv[i] = 1.0f / rowL[i + 8 * half];

  // ---- P = softmax(S) * mask, bf16 ----
  const int* mb = mask + qrow0 * KL;
  __bf16*    Pb = P    + qrow0 * KL;
#pragma unroll
  for (int kt = 0; kt < 16; ++kt) {
    const int col = key0 + kt * 16 + m16;
#pragma unroll
    for (int i = 0; i < 8; ++i) {
      const int row = i + 8 * half;
      float p = acc[kt][i] * inv[i] * (float)mb[(long long)row * KL + col];
      Pb[(long long)row * KL + col] = (__bf16)p;
    }
  }
}

// ---------------------------------------------------------------------------
extern "C" void kernel_launch(void* const* d_in, const int* in_sizes, int n_in,
                              void* d_out, int out_size, void* d_ws, size_t ws_size,
                              hipStream_t stream) {
  const float* k    = (const float*)d_in[0];  // [B, KL, E]
  const float* q    = (const float*)d_in[1];  // [B, Q, E]
  const float* W    = (const float*)d_in[2];  // [E, E]
  const int*   mask = (const int*)d_in[3];    // [B, Q, KL]
  float* out = (float*)d_out;                 // [B, Q, E]

  constexpr long long B = 16, Q = 2048, KL = 2048, E = 1024;

  __bf16* kbf = (__bf16*)d_ws;                 // 67 MB
  __bf16* Wbf = kbf + (size_t)(B * KL * E);    //  2 MB
  __bf16* qw  = Wbf + (size_t)(E * E);         // 67 MB
  __bf16* P   = qw  + (size_t)(B * Q * E);     // 134 MB

  // 0) one-time bf16 conversions of k and W
  cvt_f32_to_bf16<<<dim3((B * KL * E) / 8 / 256), 256, 0, stream>>>(k, kbf, B * KL * E);
  cvt_f32_to_bf16<<<dim3((E * E) / 8 / 256), 256, 0, stream>>>(W, Wbf, E * E);

  // 1) qw = q @ W   (flatten B*Q = 32768 rows; W shared)
  gemm_wmma_128x64<true, true>
      <<<dim3(E / 64, (B * Q) / 128, 1), 256, 0, stream>>>(
          q, Wbf, qw, (int)(B * Q), (int)E, (int)E, 0, 0, 0);

  // 2) S = qw @ k^T, softmax, * mask -> P (bf16)
  score_softmax_16q<<<dim3((B * Q) / 16), 256, 0, stream>>>(qw, kbf, mask, P);

  // 3) out = P @ k per batch
  gemm_wmma_128x64<false, false>
      <<<dim3(E / 64, Q / 128, B), 256, 0, stream>>>(
          P, kbf, out, (int)Q, (int)E, (int)KL,
          Q * KL, KL * E, Q * E);
}
